// BDEN_16621523435784
// MI455X (gfx1250) — 8-bit matrix operand order repaired and checked by reading, compiled, not run on hardware
//
#include <hip/hip_runtime.h>
#include <hip/hip_bf16.h>
#include <stdint.h>


typedef __attribute__((ext_vector_type(8))) int v8i;

#define EPSF 1e-5f

__global__ void pack_w_kernel(const float* __restrict__ w,
                              signed char* __restrict__ out,
                              int Cin, int Cout, int CoutPad, int transposed) {
  const int Ktot = 9 * Cin;
  const int total = CoutPad * Ktot;
  int idx = blockIdx.x * blockDim.x + threadIdx.x;
  if (idx >= total) return;
  int co = idx / Ktot;
  int k = idx - co * Ktot;
  int tap = k / Cin;
  int ci = k - tap * Cin;
  int kh = tap / 3;
  int kw = tap - kh * 3;
  float v = 0.0f;
  if (co < Cout) {
    if (transposed) {
      v = w[(((size_t)ci * Cout + co) * 3 + (2 - kh)) * 3 + (2 - kw)];
    } else {
      v = w[(((size_t)co * Cin + ci) * 3 + kh) * 3 + kw];
    }
  }
  out[idx] = (v > 0.0f) ? 1 : ((v < 0.0f) ? -1 : 0);
}

__global__ __launch_bounds__(256)
void conv0_kernel(const float* __restrict__ x,
                  const signed char* __restrict__ wq,
                  const float* __restrict__ bnp,
                  signed char* __restrict__ out) {
  const int Hin = 512, Win = 512, Cout = 64;
  int idx = blockIdx.x * blockDim.x + threadIdx.x;
  int co = idx & 63;
  int rest = idx >> 6;
  int ow = rest & 255; rest >>= 8;
  int oh = rest & 255;
  int n = rest >> 8;
  if (n >= 8) return;

  const signed char* wr = wq + co * 27;
  float acc = 0.0f;
#pragma unroll
  for (int kh = 0; kh < 3; ++kh) {
    int ih = oh * 2 + kh - 1;
    if (ih < 0 || ih >= Hin) continue;
#pragma unroll
    for (int kw = 0; kw < 3; ++kw) {
      int iw = ow * 2 + kw - 1;
      if (iw < 0 || iw >= Win) continue;
#pragma unroll
      for (int ci = 0; ci < 3; ++ci) {
        float wv = (float)wr[(kh * 3 + kw) * 3 + ci];
        acc += wv * x[(((size_t)n * 3 + ci) * Hin + ih) * Win + iw];
      }
    }
  }
  float g = bnp[0 * Cout + co], b = bnp[1 * Cout + co];
  float mu = bnp[2 * Cout + co], var = bnp[3 * Cout + co];
  float y = g * (acc - mu) * rsqrtf(var + EPSF) + b;
  out[idx] = (y > 0.0f) ? 1 : ((y < 0.0f) ? -1 : 0);
}

template <int CIN, int TRANSPOSED>
__global__ __launch_bounds__(256)
void binconv_wmma_kernel(const signed char* __restrict__ in,
                         const signed char* __restrict__ wq,
                         const float* __restrict__ bnp,
                         signed char* __restrict__ out_i8,
                         float* __restrict__ out_f32,
                         int Hin, int Win, int Hout, int Wout,
                         int Cout, int stride) {
  constexpr int KTOT = 9 * CIN;
  constexpr int CHUNKS = CIN / 64;

  const int lane = threadIdx.x & 31;
  const int wave = threadIdx.x >> 5;
  const unsigned m0 = blockIdx.x * 128u + wave * 16u;
  const int n0 = blockIdx.y * 32;

  const unsigned HW = (unsigned)Hout * (unsigned)Wout;
  const unsigned p = m0 + (lane & 15);
  const unsigned nb = p / HW;
  const unsigned prem = p - nb * HW;
  const int oh = (int)(prem / (unsigned)Wout);
  const int ow = (int)(prem - (unsigned)oh * (unsigned)Wout);
  const int laneHi8 = (lane >> 4) << 4;

  const int ncol = lane & 15;
  const int kHalf16 = (lane >> 4) << 4;
  const int co0 = n0 + ncol;
  const int co1 = co0 + 16;
  const signed char* w0 = wq + (size_t)co0 * KTOT + kHalf16;
  const signed char* w1 = wq + (size_t)co1 * KTOT + kHalf16;

  v8i acc0 = {};
  v8i acc1 = {};

#pragma unroll
  for (int tap = 0; tap < 9; ++tap) {
    const int kh = tap / 3;
    const int kw = tap - kh * 3;
    bool valid;
    int ih, iw;
    if (TRANSPOSED) {
      const int ihd = oh + kh - 1;
      const int iwd = ow + kw - 1;
      ih = ihd >> 1;
      iw = iwd >> 1;
      valid = (ihd >= 0) && (iwd >= 0) && !(ihd & 1) && !(iwd & 1) &&
              (ih < Hin) && (iw < Win);
    } else {
      ih = oh * stride + kh - 1;
      iw = ow * stride + kw - 1;
      valid = (ih >= 0) && (ih < Hin) && (iw >= 0) && (iw < Win);
    }
    const signed char* aptr =
        in + (((size_t)nb * Hin + ih) * Win + iw) * (size_t)CIN + laneHi8;

#pragma unroll
    for (int c = 0; c < CHUNKS; ++c) {
      v8i a = {};
      if (valid) {
#pragma unroll
        for (int v = 0; v < 8; ++v) {
          a[v] = *(const int*)(aptr + c * 64 + (((v >> 2) << 5) | ((v & 3) << 2)));
        }
      }
      v8i b0, b1;
#pragma unroll
      for (int v = 0; v < 8; ++v) {
        const int off = tap * CIN + c * 64 + (((v >> 2) << 5) | ((v & 3) << 2));
        b0[v] = *(const int*)(w0 + off);
        b1[v] = *(const int*)(w1 + off);
      }
      acc0 = __builtin_amdgcn_wmma_i32_16x16x64_iu8(true, a, true, b0, acc0,
                                                    false, false);
      acc1 = __builtin_amdgcn_wmma_i32_16x16x64_iu8(true, a, true, b1, acc1,
                                                    false, false);
    }
  }

  const int rowAdd = (lane < 16) ? 0 : 8;
  const v8i accs[2] = {acc0, acc1};
  const int cos[2] = {co0, co1};

#pragma unroll
  for (int t = 0; t < 2; ++t) {
    const int co = cos[t];
    if (co >= Cout) continue;
    const float g = bnp[0 * Cout + co];
    const float bb = bnp[1 * Cout + co];
    const float mu = bnp[2 * Cout + co];
    const float rs = rsqrtf(bnp[3 * Cout + co] + EPSF);
#pragma unroll
    for (int r = 0; r < 8; ++r) {
      const unsigned pr = m0 + (unsigned)(r + rowAdd);
      const unsigned nn = pr / HW;
      const unsigned rr = pr - nn * HW;
      const int ohh = (int)(rr / (unsigned)Wout);
      const int oww = (int)(rr - (unsigned)ohh * (unsigned)Wout);
      const float y = g * ((float)accs[t][r] - mu) * rs + bb;
      if (out_f32) {
        out_f32[(((size_t)nn * Cout + co) * Hout + ohh) * Wout + oww] = y;
      } else {
        signed char s = (y > 0.0f) ? 1 : ((y < 0.0f) ? -1 : 0);
        out_i8[((((size_t)nn * Hout + ohh) * Wout + oww) * (size_t)Cout) + co] = s;
      }
    }
  }
}

__global__ __launch_bounds__(256)
void softmax_axis0_kernel(float* __restrict__ out, int S) {
  int i = blockIdx.x * blockDim.x + threadIdx.x;
  if (i >= S) return;
  float v[8];
  float mx = -3.402823e38f;
#pragma unroll
  for (int n = 0; n < 8; ++n) {
    v[n] = out[(size_t)n * S + i];
    mx = fmaxf(mx, v[n]);
  }
  float sum = 0.0f;
#pragma unroll
  for (int n = 0; n < 8; ++n) {
    v[n] = __expf(v[n] - mx);
    sum += v[n];
  }
  const float inv = 1.0f / sum;
#pragma unroll
  for (int n = 0; n < 8; ++n) out[(size_t)n * S + i] = v[n] * inv;
}

extern "C" void kernel_launch(void* const* d_in, const int* in_sizes, int n_in,
                              void* d_out, int out_size, void* d_ws,
                              size_t ws_size, hipStream_t stream) {
  (void)in_sizes; (void)n_in; (void)out_size; (void)ws_size;
  const float* x = (const float*)d_in[0];
  const float* w[10];
  const float* bnp[10];
  for (int i = 0; i < 10; ++i) {
    w[i] = (const float*)d_in[1 + i];
    bnp[i] = (const float*)d_in[11 + i];
  }

  struct LCfg { int cin, cout, stride, tr, hin, win, hout, wout; };
  static const LCfg cfg[10] = {
      {3,   64,  2, 0, 512, 512, 256, 256},
      {64,  64,  1, 0, 256, 256, 256, 256},
      {64,  128, 2, 0, 256, 256, 128, 128},
      {128, 128, 1, 0, 128, 128, 128, 128},
      {128, 256, 1, 0, 128, 128, 128, 128},
      {256, 256, 1, 1, 128, 128, 256, 256},
      {256, 128, 1, 0, 256, 256, 256, 256},
      {128, 128, 1, 1, 256, 256, 512, 512},
      {128, 64,  1, 0, 512, 512, 512, 512},
      {64,  19,  1, 0, 512, 512, 512, 512},
  };

  char* ws = (char*)d_ws;

  size_t woff[10];
  int coutpad[10];
  size_t cur = 0;
  for (int i = 0; i < 10; ++i) {
    const int Ktot = 9 * cfg[i].cin;
    coutpad[i] = ((cfg[i].cout + 31) / 32) * 32;
    woff[i] = cur;
    size_t sz = (size_t)coutpad[i] * (size_t)Ktot;
    cur += (sz + 255) & ~(size_t)255;
  }
  signed char* wqbase = (signed char*)ws;

  signed char* buf0 = (signed char*)(ws + (size_t)4 * 1024 * 1024);
  signed char* buf1 = buf0 + (size_t)134217728;

  for (int i = 0; i < 10; ++i) {
    const int Ktot = 9 * cfg[i].cin;
    const int total = coutpad[i] * Ktot;
    pack_w_kernel<<<(total + 255) / 256, 256, 0, stream>>>(
        w[i], wqbase + woff[i], cfg[i].cin, cfg[i].cout, coutpad[i], cfg[i].tr);
  }

  {
    const int total = 8 * 256 * 256 * 64;
    conv0_kernel<<<total / 256, 256, 0, stream>>>(x, wqbase + woff[0], bnp[0],
                                                  buf0);
  }

  signed char* bufs[2] = {buf0, buf1};
  int curb = 0;
  for (int i = 1; i <= 9; ++i) {
    const LCfg& c = cfg[i];
    const unsigned P = 8u * (unsigned)c.hout * (unsigned)c.wout;
    dim3 grid(P / 128u, (unsigned)(coutpad[i] / 32));
    const signed char* src = bufs[curb];
    signed char* dst_i8 = (i == 9) ? nullptr : bufs[curb ^ 1];
    float* dst_f32 = (i == 9) ? (float*)d_out : nullptr;
    const signed char* wqi = wqbase + woff[i];

    if (c.cin == 64) {
      binconv_wmma_kernel<64, 0><<<grid, 256, 0, stream>>>(
          src, wqi, bnp[i], dst_i8, dst_f32, c.hin, c.win, c.hout, c.wout,
          c.cout, c.stride);
    } else if (c.cin == 128 && !c.tr) {
      binconv_wmma_kernel<128, 0><<<grid, 256, 0, stream>>>(
          src, wqi, bnp[i], dst_i8, dst_f32, c.hin, c.win, c.hout, c.wout,
          c.cout, c.stride);
    } else if (c.cin == 128 && c.tr) {
      binconv_wmma_kernel<128, 1><<<grid, 256, 0, stream>>>(
          src, wqi, bnp[i], dst_i8, dst_f32, c.hin, c.win, c.hout, c.wout,
          c.cout, c.stride);
    } else if (c.cin == 256 && !c.tr) {
      binconv_wmma_kernel<256, 0><<<grid, 256, 0, stream>>>(
          src, wqi, bnp[i], dst_i8, dst_f32, c.hin, c.win, c.hout, c.wout,
          c.cout, c.stride);
    } else {
      binconv_wmma_kernel<256, 1><<<grid, 256, 0, stream>>>(
          src, wqi, bnp[i], dst_i8, dst_f32, c.hin, c.win, c.hout, c.wout,
          c.cout, c.stride);
    }
    curb ^= 1;
  }

  {
    const int S = 19 * 512 * 512;
    softmax_axis0_kernel<<<(S + 255) / 256, 256, 0, stream>>>((float*)d_out, S);
  }
}
